// SimVQ_31756988187167
// MI455X (gfx1250) — compile-verified, matrix operand order checked by reading, not run on hardware
//
#include <hip/hip_runtime.h>
#include <math.h>

typedef __attribute__((ext_vector_type(16))) __bf16 v16bf;
typedef __attribute__((ext_vector_type(8)))  __bf16 v8bf;
typedef __attribute__((ext_vector_type(8)))  float  v8f;
typedef __attribute__((ext_vector_type(2)))  float  v2f;
typedef __attribute__((ext_vector_type(4)))  int    v4i;

#define EPSV 1e-6f

// Async global->LDS copy (gfx1250, tracked by ASYNCcnt). Guarded so the file
// still compiles on toolchains without the builtin (falls back to ld/st fill).
// Probe-discovered signature: (int4 addrspace(1)*, int4 addrspace(3)*, imm, imm)
#if defined(__has_builtin)
#if __has_builtin(__builtin_amdgcn_global_load_async_to_lds_b128)
#define HAVE_ASYNC_LDS 1
#endif
#endif

#ifdef HAVE_ASYNC_LDS
typedef __attribute__((address_space(1))) v4i* g4i_t;  // global int4*
typedef __attribute__((address_space(3))) v4i* l4i_t;  // LDS int4*
#endif

// ---------------------------------------------------------------------------
// Kernel 1: codebook = frozen_codebook @ weight^T   (f32 WMMA 16x16x4)
// A = F [C x D], B = W^T so B[k, d] = W[d, k]  -> B fragments are rows of W
// 32-bit A 16x4 layout: lane l (l15 = l&15, kh = l>>4): m = l15,
//   VGPR j in {0,1} holds K = kh*2 + j  -> contiguous v2f at arow + k + kh*2.
// ---------------------------------------------------------------------------
__global__ __launch_bounds__(128)
void cb_gemm_f32(const float* __restrict__ F, const float* __restrict__ W,
                 float* __restrict__ CB, int C, int D)
{
    const int lane = threadIdx.x & 31;
    const int wav  = threadIdx.x >> 5;
    const int wid  = blockIdx.x * (blockDim.x >> 5) + wav;
    const int colTiles = D >> 4;
    const int tRow = wid / colTiles;
    const int tCol = wid % colTiles;
    if (tRow * 16 >= C) return;                 // wave-uniform: EXEC stays all-1s
    const int l15 = lane & 15;
    const int kh  = lane >> 4;

    const float* arow = F + (size_t)(tRow * 16 + l15) * D;
    const float* brow = W + (size_t)(tCol * 16 + l15) * D;

    v8f acc = {};
    #pragma unroll 8
    for (int k = 0; k < D; k += 4) {
        v2f a = *(const v2f*)(arow + k + kh * 2);
        v2f b = *(const v2f*)(brow + k + kh * 2);
        acc = __builtin_amdgcn_wmma_f32_16x16x4_f32(false, a, false, b,
                                                    (short)0, acc, false, false);
    }
    // C/D layout: VGPR r -> row = tRow*16 + r + 8*kh, col = tCol*16 + l15
    float* crow = CB + (size_t)(tRow * 16 + 8 * kh) * D + tCol * 16 + l15;
    #pragma unroll
    for (int r = 0; r < 8; ++r)
        crow[(size_t)r * D] = acc[r];
}

// ---------------------------------------------------------------------------
// Kernel 2: per-code squared norm + bf16 conversion (one wave per code row)
// ---------------------------------------------------------------------------
__global__ __launch_bounds__(256)
void cb_post(const float* __restrict__ CB, __bf16* __restrict__ CBb,
             float* __restrict__ cn2, int C, int D)
{
    const int lane = threadIdx.x & 31;
    const int wav  = threadIdx.x >> 5;
    const int row  = blockIdx.x * (blockDim.x >> 5) + wav;
    if (row >= C) return;
    const float* p  = CB  + (size_t)row * D;
    __bf16*      pb = CBb + (size_t)row * D;
    float s = 0.f;
    for (int j = lane; j < D; j += 32) {
        float v = p[j];
        s += v * v;
        pb[j] = (__bf16)v;
    }
    #pragma unroll
    for (int off = 16; off; off >>= 1) s += __shfl_xor(s, off, 32);
    if (lane == 0) cn2[row] = s;
}

// ---------------------------------------------------------------------------
// Kernel 3: x -> bf16
// ---------------------------------------------------------------------------
__global__ void x_conv(const float* __restrict__ x, __bf16* __restrict__ xb,
                       size_t n)
{
    size_t i = (size_t)blockIdx.x * blockDim.x + threadIdx.x;
    if (i < n) xb[i] = (__bf16)x[i];
}

__global__ void zero1(float* p) { if (threadIdx.x == 0) *p = 0.f; }

// ---------------------------------------------------------------------------
// Kernel 4: fused score GEMM (bf16 WMMA 16x16x32) + argmin of |c|^2 - 2 x.c
//
// Block = 256 threads (8 waves), each wave owns 16 rows of x -> 128 rows/blk.
// Codebook tiles (64 codes x 256 K) are staged in LDS once per block and
// consumed by all 8 waves -> 8x reduction in L2 codebook traffic. The fill
// uses global_load_async_to_lds_b128 (ASYNCcnt) when available: no VGPR
// round-trip, completion decoupled from LOADcnt.
//
// 16-bit A 16x32 layout: lane l: m = l&15, kh = l>>4; element e of v16bf:
//   K = (e<8 ? e : 16 + (e-8)) + kh*8  -> two contiguous 16B chunks.
// B = codebook^T  =>  B fragment element = codebook[nTile + (l&15)][same K],
// read from LDS. Row stride padded to 264 elems (528B): 528/4 = 132 = 4 mod 64
// -> the 16 lanes of a b128 LDS read hit disjoint bank groups.
// ---------------------------------------------------------------------------
#define BLK_ROWS  128          // 8 waves * 16 rows
#define NTILE     64           // codes per LDS group
#define KSTAGE    256          // K elements staged per fill
#define LDS_STR   (KSTAGE + 8) // padded row stride (elements)

__device__ inline v16bf load_frag16(const __bf16* p)
{
    v8bf lo = *(const v8bf*)(p);
    v8bf hi = *(const v8bf*)(p + 16);
    return __builtin_shufflevector(lo, hi, 0, 1, 2, 3, 4, 5, 6, 7,
                                           8, 9, 10, 11, 12, 13, 14, 15);
}

__global__ __launch_bounds__(256)
void argmin_kernel(const __bf16* __restrict__ Xb, const __bf16* __restrict__ Cb,
                   const float* __restrict__ cn2, int* __restrict__ idxI,
                   float* __restrict__ idxF, int M, int C, int D)
{
    __shared__ __bf16 ldsB[NTILE * LDS_STR];    // 64 * 264 * 2B = 33,792 B

    const int tid  = threadIdx.x;
    const int lane = tid & 31;
    const int wav  = tid >> 5;
    const int l15  = lane & 15;
    const int kh   = lane >> 4;
    const int mBase = blockIdx.x * BLK_ROWS + wav * 16;
    const bool valid = (mBase < M);             // wave-uniform

    const __bf16* arow = Xb + (size_t)(mBase + l15) * D + kh * 8;

    float bestv[8];
    int   besti[8];
    #pragma unroll
    for (int r = 0; r < 8; ++r) { bestv[r] = 3.4e38f; besti[r] = 0; }

    const int totalChunks = NTILE * (KSTAGE / 8);   // 16B chunks per stage

    for (int nb = 0; nb < C; nb += NTILE) {
        v8f a0 = {}, a1 = {}, a2 = {}, a3 = {};
        for (int kc = 0; kc < D; kc += KSTAGE) {
            __syncthreads();                         // protect previous reads
            // cooperative fill: 64 rows x 256 K, 16B granules
            for (int c = tid; c < totalChunks; c += 256) {
                const int row = c >> 5;              // / (KSTAGE/8)
                const int col = (c & 31) * 8;
                const __bf16* gsrc = Cb + (size_t)(nb + row) * D + kc + col;
                __bf16*       ldst = &ldsB[row * LDS_STR + col];
#ifdef HAVE_ASYNC_LDS
                __builtin_amdgcn_global_load_async_to_lds_b128(
                    (g4i_t)(uintptr_t)gsrc,
                    (l4i_t)(unsigned int)(uintptr_t)ldst, 0, 0);
#else
                *(v8bf*)ldst = *(const v8bf*)gsrc;
#endif
            }
            // prefetch the next stage's global lines (global_prefetch_b8)
            {
                const int row = tid >> 2;            // 0..63
                const int col = (tid & 3) * 64;      // 4 x 128B per row
                int kn  = kc + KSTAGE;
                int nb2 = nb;
                if (kn >= D) { kn = 0; nb2 = (nb + NTILE < C) ? nb + NTILE : 0; }
                __builtin_prefetch(Cb + (size_t)(nb2 + row) * D + kn + col, 0, 1);
            }
#ifdef HAVE_ASYNC_LDS
            asm volatile("s_wait_asynccnt 0x0" ::: "memory");
#endif
            __syncthreads();

            if (valid) {
                const __bf16* b0 = &ldsB[(l15)      * LDS_STR + kh * 8];
                const __bf16* b1 = &ldsB[(l15 + 16) * LDS_STR + kh * 8];
                const __bf16* b2 = &ldsB[(l15 + 32) * LDS_STR + kh * 8];
                const __bf16* b3 = &ldsB[(l15 + 48) * LDS_STR + kh * 8];
                #pragma unroll 4
                for (int k = 0; k < KSTAGE; k += 32) {
                    v16bf af  = load_frag16(arow + kc + k);
                    v16bf bf0 = load_frag16(b0 + k);
                    v16bf bf1 = load_frag16(b1 + k);
                    v16bf bf2 = load_frag16(b2 + k);
                    v16bf bf3 = load_frag16(b3 + k);
                    a0 = __builtin_amdgcn_wmma_f32_16x16x32_bf16(false, af, false, bf0,
                                                                 (short)0, a0, false, false);
                    a1 = __builtin_amdgcn_wmma_f32_16x16x32_bf16(false, af, false, bf1,
                                                                 (short)0, a1, false, false);
                    a2 = __builtin_amdgcn_wmma_f32_16x16x32_bf16(false, af, false, bf2,
                                                                 (short)0, a2, false, false);
                    a3 = __builtin_amdgcn_wmma_f32_16x16x32_bf16(false, af, false, bf3,
                                                                 (short)0, a3, false, false);
                }
            }
        }
        if (valid) {
            const float c0 = cn2[nb      + l15];
            const float c1 = cn2[nb + 16 + l15];
            const float c2 = cn2[nb + 32 + l15];
            const float c3 = cn2[nb + 48 + l15];
            #pragma unroll
            for (int r = 0; r < 8; ++r) {
                float d;
                d = c0 - 2.f * a0[r]; if (d < bestv[r]) { bestv[r] = d; besti[r] = nb      + l15; }
                d = c1 - 2.f * a1[r]; if (d < bestv[r]) { bestv[r] = d; besti[r] = nb + 16 + l15; }
                d = c2 - 2.f * a2[r]; if (d < bestv[r]) { bestv[r] = d; besti[r] = nb + 32 + l15; }
                d = c3 - 2.f * a3[r]; if (d < bestv[r]) { bestv[r] = d; besti[r] = nb + 48 + l15; }
            }
        }
    }

    if (valid) {
        // Lanes sharing kh hold candidates for the same 8 rows -> 16-lane min-reduce
        #pragma unroll
        for (int off = 1; off < 16; off <<= 1) {
            #pragma unroll
            for (int r = 0; r < 8; ++r) {
                float ov = __shfl_xor(bestv[r], off, 32);
                int   oi = __shfl_xor(besti[r], off, 32);
                if (ov < bestv[r] || (ov == bestv[r] && oi < besti[r])) {
                    bestv[r] = ov; besti[r] = oi;
                }
            }
        }
        if (l15 == 0) {
            #pragma unroll
            for (int r = 0; r < 8; ++r) {
                int m = mBase + 8 * kh + r;       // VGPR r holds row r + 8*kh
                idxI[m] = besti[r];
                idxF[m] = (float)besti[r];
            }
        }
    }
}

// ---------------------------------------------------------------------------
// Kernel 5: rotation trick + commitment-loss partial (one wave per token)
// ---------------------------------------------------------------------------
__global__ __launch_bounds__(256)
void rotate_kernel(const float* __restrict__ X, const float* __restrict__ CB,
                   const int* __restrict__ idx, float* __restrict__ out,
                   float* __restrict__ lossAcc, int M, int D)
{
    const int lane = threadIdx.x & 31;
    const int wav  = threadIdx.x >> 5;
    const int tok  = blockIdx.x * (blockDim.x >> 5) + wav;
    if (tok >= M) return;
    const int NE = D >> 5;                     // 16 for D=512
    const float* xp = X  + (size_t)tok * D;
    const float* qp = CB + (size_t)idx[tok] * D;

    float xv[16], qv[16], uw[16];
    float sx = 0.f, sq = 0.f, se = 0.f;
    for (int j = 0; j < NE; ++j) {
        float xe = xp[lane + 32 * j];
        float qe = qp[lane + 32 * j];
        xv[j] = xe; qv[j] = qe;
        sx += xe * xe;
        sq += qe * qe;
        float dl = xe - qe;
        se += dl * dl;
    }
    #pragma unroll
    for (int off = 16; off; off >>= 1) {
        sx += __shfl_xor(sx, off, 32);
        sq += __shfl_xor(sq, off, 32);
        se += __shfl_xor(se, off, 32);
    }
    float nx  = sqrtf(sx);
    float nq  = sqrtf(sq);
    float rnx = 1.f / fmaxf(nx, EPSV);
    float rnq = 1.f / fmaxf(nq, EPSV);

    float sw = 0.f;
    for (int j = 0; j < NE; ++j) {
        float t = xv[j] * rnx + qv[j] * rnq;   // u + q
        uw[j] = t;
        sw += t * t;
    }
    #pragma unroll
    for (int off = 16; off; off >>= 1) sw += __shfl_xor(sw, off, 32);
    float rnw = 1.f / fmaxf(sqrtf(sw), EPSV);

    float ew = 0.f, eu = 0.f;
    for (int j = 0; j < NE; ++j) {
        ew += xv[j] * (uw[j] * rnw);
        eu += xv[j] * (xv[j] * rnx);
    }
    #pragma unroll
    for (int off = 16; off; off >>= 1) {
        ew += __shfl_xor(ew, off, 32);
        eu += __shfl_xor(eu, off, 32);
    }
    float scale = nq * rnx;                    // safe_div(nq, nx)
    float* op = out + (size_t)tok * D;
    for (int j = 0; j < NE; ++j) {
        float w = uw[j] * rnw;
        float q = qv[j] * rnq;
        op[lane + 32 * j] = (xv[j] - 2.f * ew * w + 2.f * eu * q) * scale;
    }
    if (lane == 0) atomicAdd(lossAcc, se);
}

__global__ void finalize(const float* __restrict__ acc, float* __restrict__ out,
                         float inv)
{
    if (threadIdx.x == 0) out[0] = 1.25f * acc[0] * inv;
}

// ---------------------------------------------------------------------------
extern "C" void kernel_launch(void* const* d_in, const int* in_sizes, int n_in,
                              void* d_out, int out_size, void* d_ws, size_t ws_size,
                              hipStream_t stream)
{
    const float* x  = (const float*)d_in[0];
    const float* fc = (const float*)d_in[1];
    const float* w  = (const float*)d_in[2];

    const int D = (int)(sqrt((double)in_sizes[2]) + 0.5);   // 512
    const int C = in_sizes[1] / D;                          // 4096
    const int M = in_sizes[0] / D;                          // 8192

    // workspace carve-out
    char*  base = (char*)d_ws;
    size_t off  = 0;
    auto carve = [&](size_t bytes) -> void* {
        void* p = base + off;
        off = (off + bytes + 255) & ~(size_t)255;
        return p;
    };
    float*  cbF     = (float*)carve((size_t)C * D * sizeof(float));
    __bf16* cbB     = (__bf16*)carve((size_t)C * D * sizeof(__bf16));
    __bf16* xB      = (__bf16*)carve((size_t)M * D * sizeof(__bf16));
    float*  cn2     = (float*)carve((size_t)C * sizeof(float));
    int*    idxI    = (int*)carve((size_t)M * sizeof(int));
    float*  lossAcc = (float*)carve(sizeof(float));
    (void)ws_size; (void)n_in; (void)out_size;

    float* outRot  = (float*)d_out;
    float* outIdx  = outRot + (size_t)M * D;
    float* outLoss = outIdx + M;

    // 1. codebook GEMM (f32 WMMA)
    {
        int waves  = (C / 16) * (D / 16);
        int blocks = (waves + 3) / 4;
        cb_gemm_f32<<<blocks, 128, 0, stream>>>(fc, w, cbF, C, D);
    }
    // 2. norms + bf16 codebook
    cb_post<<<(C + 7) / 8, 256, 0, stream>>>(cbF, cbB, cn2, C, D);
    // 3. x -> bf16
    {
        size_t n = (size_t)M * D;
        x_conv<<<(int)((n + 255) / 256), 256, 0, stream>>>(x, xB, n);
    }
    // 4. zero loss accumulator
    zero1<<<1, 32, 0, stream>>>(lossAcc);
    // 5. fused score GEMM + argmin (bf16 WMMA, async-LDS-tiled codebook)
    {
        int blocks = (M + BLK_ROWS - 1) / BLK_ROWS;
        argmin_kernel<<<blocks, 256, 0, stream>>>(xB, cbB, cn2, idxI, outIdx, M, C, D);
    }
    // 6. rotation trick + loss partials
    rotate_kernel<<<(M + 7) / 8, 256, 0, stream>>>(x, cbF, idxI, outRot, lossAcc, M, D);
    // 7. finalize scalar loss
    finalize<<<1, 32, 0, stream>>>(lossAcc, outLoss, 1.f / ((float)M * (float)D));
}
